// GraphNetwork_83004537962777
// MI455X (gfx1250) — matrix operand order repaired and checked by reading, not compiled, not run on hardware
//
#include <hip/hip_runtime.h>

typedef __attribute__((ext_vector_type(16))) __bf16 v16bf;
typedef __attribute__((ext_vector_type(8)))  float  v8f;
typedef __attribute__((ext_vector_type(4)))  unsigned v4u;
typedef __attribute__((ext_vector_type(4)))  int v4i;
typedef __attribute__((ext_vector_type(8)))  int v8i;

#define Nn   20000
#define Ee   640000
#define INd  256
#define OUTd 800
#define OUTP 832       // padded cols (multiple of BN), rows 800..831 zero
#define Rr   8
#define Bb   8
#define KS   2048      // R*IN (s part of K)
#define Ktot 2304      // R*IN + IN
#define K2   (Ktot/2)  // 1152 K-pairs

#define BM   128
#define BN   64
#define KC   64
#define NCH  (Ktot / KC)   // 36 chunks, boundary 2048 is chunk-aligned
#define KCP2 68        // A row stride in LDS (64 dwords + TDM pad 4)
#define BK2  36        // B row stride in LDS (32 dwords + TDM pad 4)

// ---- one-time RNE f32->bf16 split (used only in build_w) ----
static __device__ __forceinline__ unsigned short f2bf(float f) {
  unsigned u = __float_as_uint(f);
  unsigned r = u + 0x7FFFu + ((u >> 16) & 1u);
  return (unsigned short)(r >> 16);
}
static __device__ __forceinline__ float bf2f(unsigned short h) {
  return __uint_as_float(((unsigned)h) << 16);
}
static __device__ __forceinline__ void splitpack_rne(float a, float b,
                                                     unsigned& hi, unsigned& lo) {
  unsigned short ha = f2bf(a), hb = f2bf(b);
  unsigned short la = f2bf(a - bf2f(ha)), lb = f2bf(b - bf2f(hb));
  hi = ((unsigned)hb << 16) | (unsigned)ha;
  lo = ((unsigned)lb << 16) | (unsigned)la;
}

// ---- hot-loop truncation split: lowers to v_perm + v_fma_mix per pair ----
static __device__ __forceinline__ void splitpack_t(float a, float b,
                                                   unsigned& hi, unsigned& lo) {
  unsigned au = __float_as_uint(a), bu = __float_as_uint(b);
  hi = __builtin_amdgcn_perm(bu, au, 0x07060302u);        // {b_hi16, a_hi16}
  float ar = a - __uint_as_float(au & 0xffff0000u);
  float br = b - __uint_as_float(bu & 0xffff0000u);
  lo = __builtin_amdgcn_perm(__float_as_uint(br), __float_as_uint(ar), 0x07060302u);
}

// ---- Tensor Data Mover: one 2-D tile load, descriptor per ISA cdna5 §8.3/8.4 ----
// td0/td1 = remaining extent from tile start (OOB -> zero-fill), units = dwords.
static __device__ __forceinline__ void tdm_load_2d(
    unsigned lds_byte, const void* gptr,
    unsigned tile0, unsigned tile1,
    unsigned td0, unsigned td1,
    unsigned long stride0,
    unsigned pad_interval_code, unsigned pad_amount_code) {
  unsigned long ga = (unsigned long)gptr;
  v4u g0;
  g0.x = 1u;                                               // count=1 user D#
  g0.y = lds_byte;                                         // lds_addr[31:0]
  g0.z = (unsigned)ga;                                     // global_addr[31:0]
  g0.w = ((unsigned)(ga >> 32) & 0x01FFFFFFu) | 0x80000000u; // ga[56:32] | type=2
  v8i g1;
  g1[0] = (int)((2u << 16) | (1u << 20) |                  // data_size=4B, pad_en
                (pad_interval_code << 22) | (pad_amount_code << 25));
  g1[1] = (int)((td0 & 0xffffu) << 16);                    // atomic_bar=0 | td0.lo
  g1[2] = (int)((td0 >> 16) | ((td1 & 0xffffu) << 16));    // td0.hi | td1.lo
  g1[3] = (int)((td1 >> 16) | (tile0 << 16));              // td1.hi | tile_dim0
  g1[4] = (int)tile1;                                      // tile_dim1 | tile_dim2=0
  g1[5] = (int)(unsigned)stride0;                          // dim0_stride[31:0]
  g1[6] = (int)((unsigned)(stride0 >> 32) & 0xffffu);      // dim0_stride[47:32]
  g1[7] = 0;                                               // dim1_stride (unused, 2D)
  v4i z4 = {0, 0, 0, 0};
#if defined(__clang_major__) && __clang_major__ >= 23
  v8i z8 = {0, 0, 0, 0, 0, 0, 0, 0};
  __builtin_amdgcn_tensor_load_to_lds(g0, g1, z4, z4, z8, 0);
#else
  __builtin_amdgcn_tensor_load_to_lds(g0, g1, z4, z4, 0);
#endif
}

// ---------------- stage 0: zero the segment-sum scratch -----------------
__global__ void zero_f4(float4* __restrict__ p, long n4) {
  long i = (long)blockIdx.x * blockDim.x + threadIdx.x;
  long stride = (long)gridDim.x * blockDim.x;
  float4 z; z.x = 0.f; z.y = 0.f; z.z = 0.f; z.w = 0.f;
  for (; i < n4; i += stride) p[i] = z;
}

// ---- stage 1: Wt[n][k2] hi/lo bf16-pair weights, pre-transposed + padded ----
static __device__ __forceinline__ float wval(int k, int o,
                                             const float* __restrict__ att,
                                             const float* __restrict__ bases,
                                             const float* __restrict__ root) {
  if (k < KS) {
    int r = k >> 8, i = k & 255;
    float v = 0.f;
#pragma unroll
    for (int b = 0; b < Bb; ++b)
      v += att[r * Bb + b] * bases[((long)b * INd + i) * OUTd + o];
    return v;
  }
  return root[(long)(k - KS) * OUTd + o];
}

__global__ void build_w(const float* __restrict__ att, const float* __restrict__ bases,
                        const float* __restrict__ root,
                        unsigned* __restrict__ WtHi, unsigned* __restrict__ WtLo) {
  int idx = blockIdx.x * 256 + threadIdx.x;   // o * K2 + k2
  if (idx >= OUTP * K2) return;
  int o = idx / K2, k2 = idx % K2;
  float v0 = 0.f, v1 = 0.f;
  if (o < OUTd) {
    v0 = wval(2 * k2,     o, att, bases, root);
    v1 = wval(2 * k2 + 1, o, att, bases, root);
  }
  unsigned hi, lo;
  splitpack_rne(v0, v1, hi, lo);
  WtHi[idx] = hi;
  WtLo[idx] = lo;
}

// ---------------- stage 2: edge scatter-add into s[N][R][IN] ----------------
__global__ void scatter_add(const float* __restrict__ x, const int* __restrict__ src,
                            const int* __restrict__ dst, const int* __restrict__ etype,
                            float* __restrict__ s) {
  int e = blockIdx.x;
  int u = src[e], v = dst[e], r = etype[e];
  const float* xr = x + (long)u * INd;
  float* sr = s + ((long)v * Rr + r) * INd;
  atomicAdd(&sr[threadIdx.x], xr[threadIdx.x]);
}

// ---------------- stage 3: TDM-staged split-bf16 WMMA GEMM ----------------
__global__ __launch_bounds__(256) void rgcn_gemm(
    const float* __restrict__ s, const float* __restrict__ x,
    const unsigned* __restrict__ WtHi, const unsigned* __restrict__ WtLo,
    const float* __restrict__ bias, float* __restrict__ out) {
  __shared__ alignas(16) float        As[2][BM * KCP2];   // f32 A tile (TDM-padded)
  __shared__ alignas(16) unsigned int BsHi[2][BN * BK2];  // Bs[n][k2] hi pairs
  __shared__ alignas(16) unsigned int BsLo[2][BN * BK2];

  const int tid  = threadIdx.x;
  const int lane = tid & 31;
  const int wave = tid >> 5;        // 8 row tiles of 16 rows
  const int m0w  = wave * 16;
  const int mrow = lane & 15;
  const int h    = lane >> 4;       // lane half selects K sub-block

  const int block_m = blockIdx.x * BM;
  const int block_n = blockIdx.y * BN;

  v8f acc[4] = {{}, {}, {}, {}};

  // ---- TDM stage of one K-chunk into LDS buffer `buf` (3 descriptors) ----
  auto stage = [&](int c, int buf) {
    const int k0 = c * KC;
    const float* abase; unsigned astride; int akoff;
    if (k0 < KS) { abase = s; astride = KS;  akoff = k0;      }
    else         { abase = x; astride = INd; akoff = k0 - KS; }
    // A: 128 rows x 64 dwords, LDS row stride 68 via pad(64dw interval, 4dw)
    tdm_load_2d((unsigned)(size_t)(&As[buf][0]),
                abase + (long)block_m * astride + akoff,
                /*tile*/ KC, BM,
                /*remaining*/ astride - (unsigned)akoff, (unsigned)(Nn - block_m),
                /*stride*/ astride, /*pad codes*/ 5u, 3u);
    // B hi/lo: 64 rows x 32 dwords, LDS row stride 36 via pad(32dw, 4dw)
    long goff = (long)block_n * K2 + (k0 >> 1);
    tdm_load_2d((unsigned)(size_t)(&BsHi[buf][0]), WtHi + goff,
                32u, BN, (unsigned)(K2 - (k0 >> 1)), (unsigned)(OUTP - block_n),
                (unsigned long)K2, 4u, 3u);
    tdm_load_2d((unsigned)(size_t)(&BsLo[buf][0]), WtLo + goff,
                32u, BN, (unsigned)(K2 - (k0 >> 1)), (unsigned)(OUTP - block_n),
                (unsigned long)K2, 4u, 3u);
  };

  if (wave == 0) stage(0, 0);

  for (int c = 0; c < NCH; ++c) {
    const int cur = c & 1;
    if (wave == 0) {
      if (c + 1 < NCH) {
        stage(c + 1, cur ^ 1);
        __builtin_amdgcn_s_wait_tensorcnt(3);   // chunk c landed, c+1 in flight
      } else {
        __builtin_amdgcn_s_wait_tensorcnt(0);
      }
    }
    __syncthreads();

#pragma unroll
    for (int ks = 0; ks < KC; ks += 32) {
      // A fragment: lane half h covers K = ks+8h..+7 and ks+16+8h..+7 (f32->split)
      const float* ap = &As[cur][(m0w + mrow) * KCP2 + ks + 8 * h];
      float4 f0 = *(const float4*)(ap);
      float4 f1 = *(const float4*)(ap + 4);
      float4 f2 = *(const float4*)(ap + 16);
      float4 f3 = *(const float4*)(ap + 20);
      union { unsigned u[8]; v16bf v; } aHi, aLo;
      splitpack_t(f0.x, f0.y, aHi.u[0], aLo.u[0]);
      splitpack_t(f0.z, f0.w, aHi.u[1], aLo.u[1]);
      splitpack_t(f1.x, f1.y, aHi.u[2], aLo.u[2]);
      splitpack_t(f1.z, f1.w, aHi.u[3], aLo.u[3]);
      splitpack_t(f2.x, f2.y, aHi.u[4], aLo.u[4]);
      splitpack_t(f2.z, f2.w, aHi.u[5], aLo.u[5]);
      splitpack_t(f3.x, f3.y, aHi.u[6], aLo.u[6]);
      splitpack_t(f3.z, f3.w, aHi.u[7], aLo.u[7]);

#pragma unroll
      for (int j = 0; j < 4; ++j) {        // one A split feeds 12 WMMAs
        int ncol = j * 16 + mrow;
        const unsigned* bp = &BsHi[cur][ncol * BK2 + (ks >> 1) + 4 * h];
        const unsigned* bq = &BsLo[cur][ncol * BK2 + (ks >> 1) + 4 * h];
        union { uint4 q[2]; v16bf v; } bHi, bLo;
        bHi.q[0] = *(const uint4*)(bp);  bHi.q[1] = *(const uint4*)(bp + 8);
        bLo.q[0] = *(const uint4*)(bq);  bLo.q[1] = *(const uint4*)(bq + 8);

        v8f a = acc[j];
        a = __builtin_amdgcn_wmma_f32_16x16x32_bf16(false, aHi.v, false, bHi.v, (short)0, a, false, false);
        a = __builtin_amdgcn_wmma_f32_16x16x32_bf16(false, aLo.v, false, bHi.v, (short)0, a, false, false);
        a = __builtin_amdgcn_wmma_f32_16x16x32_bf16(false, aHi.v, false, bLo.v, (short)0, a, false, false);
        acc[j] = a;
      }
    }
    __syncthreads();   // all waves done reading buf before it is re-staged
  }

  // epilogue: D layout — VGPR q, lanes0-15 -> M=q, lanes16-31 -> M=q+8
#pragma unroll
  for (int j = 0; j < 4; ++j) {
    v8f a = acc[j];
    int col = block_n + j * 16 + mrow;
    if (col < OUTd) {
      float bv = bias[col];
#pragma unroll
      for (int q = 0; q < 8; ++q) {
        int row = block_m + m0w + q + 8 * h;
        if (row < Nn) out[(long)row * OUTd + col] = a[q] + bv;
      }
    }
  }
}

extern "C" void kernel_launch(void* const* d_in, const int* in_sizes, int n_in,
                              void* d_out, int out_size, void* d_ws, size_t ws_size,
                              hipStream_t stream) {
  const float* x     = (const float*)d_in[0];
  const int*   eidx  = (const int*)  d_in[1];
  const int*   etype = (const int*)  d_in[2];
  const float* bases = (const float*)d_in[3];
  const float* att   = (const float*)d_in[4];
  const float* root  = (const float*)d_in[5];
  const float* bias  = (const float*)d_in[6];
  float* out = (float*)d_out;

  float*    s    = (float*)d_ws;                          // [N,R,IN] 40,960,000 f32
  unsigned* WtHi = (unsigned*)(s + (size_t)Nn * Rr * INd);// [832][1152] uint
  unsigned* WtLo = WtHi + (size_t)OUTP * K2;

  const int* src = eidx;        // edge_index[0]
  const int* dst = eidx + Ee;   // edge_index[1]

  zero_f4<<<4096, 256, 0, stream>>>((float4*)s, (long)Nn * Rr * INd / 4);
  build_w<<<(OUTP * K2 + 255) / 256, 256, 0, stream>>>(att, bases, root, WtHi, WtLo);
  scatter_add<<<Ee, INd, 0, stream>>>(x, src, dst, etype, s);
  rgcn_gemm<<<dim3((Nn + BM - 1) / BM, (OUTd + BN - 1) / BN), 256, 0, stream>>>(
      s, x, WtHi, WtLo, bias, out);
}
